// AttentionSumPooling_60249801228695
// MI455X (gfx1250) — matrix operand order repaired and checked by reading, not compiled, not run on hardware
//
#include <hip/hip_runtime.h>
#include <hip/hip_bf16.h>

// ---------------- problem constants (from reference) ----------------
#define N_NODES   20000
#define N_EDGES   320000
#define IN_DIM    128
#define HID       256
#define HEADS     8
#define DH        32
#define LAYERS    3
#define NUM_GRAPHS 64
#define OUT_DIM   10
#define SLOPE     0.2f

// ---------------- WMMA vector types (CDNA5 / gfx1250) ----------------
typedef __attribute__((ext_vector_type(16))) _Float16 v16h_t;
typedef __attribute__((ext_vector_type(8)))  _Float16 v8h_t;
typedef __attribute__((ext_vector_type(8)))  float    v8f_t;

union V16H { v16h_t v; v8h_t h[2]; };

// =====================================================================
// WMMA GEMM:  C[Mtot x 256] = A[Mtot x K](f16) @ B[K x 256] + bias
// B is passed pre-transposed: Bt[256 x K] f16 row-major, so each lane's
// B fragment is two contiguous 8-value K runs of one output column.
// Block = 512 threads = 16 waves; wave w owns column tile w (16 cols).
// A row-panel (16 x K) is staged in LDS once and reused by all 16 waves.
//
// Fragment layouts per CDNA5 ISA 7.12.2 (wave32):
//   A 16x32 f16 : lane L -> row m=L%16, g=L/16; elems 0-7 = K 8g..8g+7,
//                 elems 8-15 = K 16+8g..16+8g+7  (two contiguous v8h)
//   B 32x16 f16 : lane L -> col n=L%16, g=L/16; elems 0-7 = K 8g..8g+7, 8-15 = K 16+8g..16+8g+7
//                 (two contiguous v8h from Bt row n)
//   C/D 16x16 f32: VGPR r, lane L -> row r+8g, col n=L%16
// =====================================================================
__global__ __launch_bounds__(512) void gat_gemm_wmma(
    const _Float16* __restrict__ A,    // [Mtot x K] f16 row-major
    const _Float16* __restrict__ Bt,   // [256 x K]  f16 row-major (B^T)
    const float*    __restrict__ bias, // [256]
    float*          __restrict__ C,    // [Mtot x 256] f32
    int K)
{
    __shared__ __align__(16) _Float16 As[16 * 256];  // 8 KB panel, stride 256

    const int tid  = threadIdx.x;
    const int row0 = blockIdx.x * 16;

    // stage A panel -> LDS (b128 chunks)
    const int chunksPerRow = K >> 3;                 // v8h chunks per row
    for (int i = tid; i < 16 * chunksPerRow; i += 512) {
        const int r = i / chunksPerRow;
        const int c = i - r * chunksPerRow;
        ((v8h_t*)As)[r * 32 + c] =
            *(const v8h_t*)(A + (size_t)(row0 + r) * K + c * 8);
    }
    __syncthreads();

    const int wave = tid >> 5;          // column tile 0..15
    const int lane = tid & 31;
    const int mn   = lane & 15;         // row-in-tile (A) / col-in-tile (B)
    const int g    = lane >> 4;

    const _Float16* arow = As + mn * 256;
    const _Float16* brow = Bt + (size_t)(wave * 16 + mn) * K;

    v8f_t acc = {};
    for (int k0 = 0; k0 < K; k0 += 32) {
        V16H a, b;
        a.h[0] = *(const v8h_t*)(arow + k0 + 8 * g);
        a.h[1] = *(const v8h_t*)(arow + k0 + 16 + 8 * g);
        b.h[0] = *(const v8h_t*)(brow + k0 + 8 * g);
        b.h[1] = *(const v8h_t*)(brow + k0 + 16 + 8 * g);
        acc = __builtin_amdgcn_wmma_f32_16x16x32_f16(
            /*neg_a=*/false, a.v, /*neg_b=*/false, b.v,
            /*c_mod=*/(short)0, acc, /*reuse_a=*/false, /*reuse_b=*/false);
    }

    const int   col0 = wave * 16;
    const float bv   = bias[col0 + mn];
#pragma unroll
    for (int r = 0; r < 8; ++r)
        C[(size_t)(row0 + r + 8 * g) * 256 + col0 + mn] = acc[r] + bv;
}

// ---------------- helpers / elementwise kernels ----------------------

__device__ __forceinline__ void atomicMaxF(float* addr, float v) {
    // sign-punned f32 max: positive floats order as ints, negatives reverse as uints
    if (v >= 0.f) atomicMax((int*)addr, __float_as_int(v));
    else          atomicMin((unsigned int*)addr, __float_as_uint(v));
}

__global__ void gat_wt_f16(const float* __restrict__ W,  // [K x M] f32
                           _Float16* __restrict__ Wt,    // [M x K] f16
                           int K, int M)
{
    const int t = blockIdx.x * blockDim.x + threadIdx.x;
    if (t >= K * M) return;
    const int k = t / M, m = t - k * M;
    Wt[(size_t)m * K + k] = (_Float16)W[t];
}

__global__ void gat_f16(const float* __restrict__ x, _Float16* __restrict__ y, int n)
{
    const int i = blockIdx.x * blockDim.x + threadIdx.x;
    if (i < n) y[i] = (_Float16)x[i];
}

__global__ void gat_init_layer(float* __restrict__ rst,
                               float* __restrict__ nmax,
                               float* __restrict__ denom)
{
    const int i = blockIdx.x * blockDim.x + threadIdx.x;
    if (i < N_NODES * HID) rst[i] = 0.f;
    if (i < N_NODES * HEADS) { nmax[i] = -__builtin_inff(); denom[i] = 0.f; }
}

// score_{e,h} = a_h . leakyrelu(fs[src]_h + fd[dst]_h);  segment-max into nmax
__global__ void gat_edge_score(const float* __restrict__ fs,
                               const float* __restrict__ fd,
                               const float* __restrict__ attn,   // [HEADS x DH]
                               const int*   __restrict__ src,
                               const int*   __restrict__ dst,
                               float*       __restrict__ scores, // [E x HEADS]
                               float*       __restrict__ nmax)
{
    const int t = blockIdx.x * blockDim.x + threadIdx.x;
    if (t >= N_EDGES * HEADS) return;
    const int e = t >> 3, hh = t & 7;
    const int s = src[e], d = dst[e];
    const float4* ps = (const float4*)(fs + (size_t)s * HID + hh * DH);
    const float4* pd = (const float4*)(fd + (size_t)d * HID + hh * DH);
    const float4* pa = (const float4*)(attn + hh * DH);
    float sc = 0.f;
#pragma unroll
    for (int i = 0; i < DH / 4; ++i) {
        const float4 a4 = pa[i], s4 = ps[i], d4 = pd[i];
        float v;
        v = s4.x + d4.x; sc += a4.x * (v > 0.f ? v : SLOPE * v);
        v = s4.y + d4.y; sc += a4.y * (v > 0.f ? v : SLOPE * v);
        v = s4.z + d4.z; sc += a4.z * (v > 0.f ? v : SLOPE * v);
        v = s4.w + d4.w; sc += a4.w * (v > 0.f ? v : SLOPE * v);
    }
    scores[t] = sc;
    atomicMaxF(&nmax[(size_t)d * HEADS + hh], sc);
}

__global__ void gat_edge_exp(const int* __restrict__ dst,
                             float* __restrict__ scores,
                             const float* __restrict__ nmax,
                             float* __restrict__ denom)
{
    const int t = blockIdx.x * blockDim.x + threadIdx.x;
    if (t >= N_EDGES * HEADS) return;
    const int e = t >> 3, hh = t & 7;
    const int d = dst[e];
    const float ex = expf(scores[t] - nmax[(size_t)d * HEADS + hh]);
    scores[t] = ex;
    atomicAdd(&denom[(size_t)d * HEADS + hh], ex);
}

// one block per edge, 256 threads = 256 channels: rst[dst] += alpha * fs[src]
__global__ __launch_bounds__(256) void gat_edge_scatter(
    const float* __restrict__ fs,
    const int*   __restrict__ src,
    const int*   __restrict__ dst,
    const float* __restrict__ scores,
    const float* __restrict__ denom,
    float*       __restrict__ rst)
{
    const int e = blockIdx.x, c = threadIdx.x, hh = c >> 5;
    const int s = src[e], d = dst[e];
    const float alpha = scores[(size_t)e * HEADS + hh] /
                        denom[(size_t)d * HEADS + hh];
    atomicAdd(&rst[(size_t)d * HID + c], alpha * fs[(size_t)s * HID + c]);
}

__global__ void gat_relu_res(float* __restrict__ h, const float* __restrict__ rst)
{
    const int i = blockIdx.x * blockDim.x + threadIdx.x;
    if (i >= N_NODES * HID) return;
    const float v = rst[i] + h[i];
    h[i] = v > 0.f ? v : 0.f;
}

__global__ void gat_pool_init(float* __restrict__ hg)
{
    const int i = blockIdx.x * blockDim.x + threadIdx.x;
    if (i < NUM_GRAPHS * HID) hg[i] = 0.f;
}

__global__ __launch_bounds__(256) void gat_pool(const float* __restrict__ h,
                                                const int* __restrict__ gid,
                                                float* __restrict__ hg)
{
    const int node = blockIdx.x, c = threadIdx.x;
    atomicAdd(&hg[(size_t)gid[node] * HID + c], h[(size_t)node * HID + c]);
}

__global__ void gat_fc(const float* __restrict__ X,  // [G x Kd]
                       const float* __restrict__ W,  // [Kd x Nd] row-major
                       const float* __restrict__ b,  // [Nd]
                       float*       __restrict__ Y,  // [G x Nd]
                       int Kd, int Nd, int doRelu)
{
    const int t = blockIdx.x * blockDim.x + threadIdx.x;
    if (t >= NUM_GRAPHS * Nd) return;
    const int g = t / Nd, j = t - g * Nd;
    float acc = b[j];
    for (int k = 0; k < Kd; ++k)
        acc += X[(size_t)g * Kd + k] * W[(size_t)k * Nd + j];
    Y[t] = doRelu ? (acc > 0.f ? acc : 0.f) : acc;
}

// =====================================================================
extern "C" void kernel_launch(void* const* d_in, const int* in_sizes, int n_in,
                              void* d_out, int out_size, void* d_ws, size_t ws_size,
                              hipStream_t stream)
{
    (void)in_sizes; (void)n_in; (void)out_size; (void)ws_size;

    const float* feature = (const float*)d_in[0];
    const float* W_in    = (const float*)d_in[1];
    const float* b_in    = (const float*)d_in[2];
    const float* W_src   = (const float*)d_in[3];
    const float* b_src   = (const float*)d_in[4];
    const float* W_dst   = (const float*)d_in[5];
    const float* b_dst   = (const float*)d_in[6];
    const float* attn    = (const float*)d_in[7];
    const float* Wc1     = (const float*)d_in[8];
    const float* bc1     = (const float*)d_in[9];
    const float* Wc2     = (const float*)d_in[10];
    const float* bc2     = (const float*)d_in[11];
    const float* Wc3     = (const float*)d_in[12];
    const float* bc3     = (const float*)d_in[13];
    const int*   src     = (const int*)d_in[14];
    const int*   dst     = (const int*)d_in[15];
    const int*   gid     = (const int*)d_in[16];
    float*       out     = (float*)d_out;

    // -------- workspace suballocation (256B aligned), ~105 MB total ----
    char* p = (char*)d_ws;
    auto alloc = [&](size_t bytes) -> char* {
        char* r = p; p += (bytes + 255) & ~(size_t)255; return r;
    };
    float*    h      = (float*)   alloc((size_t)N_NODES * HID * 4);
    float*    fs     = (float*)   alloc((size_t)N_NODES * HID * 4);
    float*    fd     = (float*)   alloc((size_t)N_NODES * HID * 4);
    float*    rst    = (float*)   alloc((size_t)N_NODES * HID * 4);
    float*    scores = (float*)   alloc((size_t)N_EDGES * HEADS * 4);
    float*    nmax   = (float*)   alloc((size_t)N_NODES * HEADS * 4);
    float*    denom  = (float*)   alloc((size_t)N_NODES * HEADS * 4);
    _Float16* a16    = (_Float16*)alloc((size_t)N_NODES * HID * 2); // also feat16
    _Float16* WinT   = (_Float16*)alloc((size_t)HID * IN_DIM * 2);
    _Float16* WsT    = (_Float16*)alloc((size_t)LAYERS * HID * HID * 2);
    _Float16* WdT    = (_Float16*)alloc((size_t)LAYERS * HID * HID * 2);
    float*    hg     = (float*)   alloc((size_t)NUM_GRAPHS * HID * 4);
    float*    x1     = (float*)   alloc((size_t)NUM_GRAPHS * HID * 4);
    float*    x2     = (float*)   alloc((size_t)NUM_GRAPHS * (HID / 2) * 4);

    const int B = 256;

    // -------- one-time weight convert+transpose to f16 -----------------
    gat_wt_f16<<<(IN_DIM * HID + B - 1) / B, B, 0, stream>>>(W_in, WinT, IN_DIM, HID);
    for (int l = 0; l < LAYERS; ++l) {
        gat_wt_f16<<<(HID * HID + B - 1) / B, B, 0, stream>>>(
            W_src + (size_t)l * HID * HID, WsT + (size_t)l * HID * HID, HID, HID);
        gat_wt_f16<<<(HID * HID + B - 1) / B, B, 0, stream>>>(
            W_dst + (size_t)l * HID * HID, WdT + (size_t)l * HID * HID, HID, HID);
    }

    // -------- input projection: h = feature @ W_in + b_in --------------
    gat_f16<<<(N_NODES * IN_DIM + B - 1) / B, B, 0, stream>>>(feature, a16, N_NODES * IN_DIM);
    gat_gemm_wmma<<<N_NODES / 16, 512, 0, stream>>>(a16, WinT, b_in, h, IN_DIM);

    // -------- 3 GATv2 layers -------------------------------------------
    for (int l = 0; l < LAYERS; ++l) {
        gat_f16<<<(N_NODES * HID + B - 1) / B, B, 0, stream>>>(h, a16, N_NODES * HID);
        gat_gemm_wmma<<<N_NODES / 16, 512, 0, stream>>>(
            a16, WsT + (size_t)l * HID * HID, b_src + l * HID, fs, HID);
        gat_gemm_wmma<<<N_NODES / 16, 512, 0, stream>>>(
            a16, WdT + (size_t)l * HID * HID, b_dst + l * HID, fd, HID);

        gat_init_layer<<<(N_NODES * HID + B - 1) / B, B, 0, stream>>>(rst, nmax, denom);

        const int et = N_EDGES * HEADS;
        gat_edge_score<<<(et + B - 1) / B, B, 0, stream>>>(
            fs, fd, attn + (size_t)l * HEADS * DH, src, dst, scores, nmax);
        gat_edge_exp<<<(et + B - 1) / B, B, 0, stream>>>(dst, scores, nmax, denom);
        gat_edge_scatter<<<N_EDGES, 256, 0, stream>>>(fs, src, dst, scores, denom, rst);
        gat_relu_res<<<(N_NODES * HID + B - 1) / B, B, 0, stream>>>(h, rst);
    }

    // -------- readout + classifier -------------------------------------
    gat_pool_init<<<(NUM_GRAPHS * HID + B - 1) / B, B, 0, stream>>>(hg);
    gat_pool<<<N_NODES, 256, 0, stream>>>(h, gid, hg);

    gat_fc<<<(NUM_GRAPHS * HID + B - 1) / B, B, 0, stream>>>(hg, Wc1, bc1, x1, HID, HID, 1);
    gat_fc<<<(NUM_GRAPHS * (HID / 2) + B - 1) / B, B, 0, stream>>>(x1, Wc2, bc2, x2, HID, HID / 2, 1);
    gat_fc<<<(NUM_GRAPHS * OUT_DIM + B - 1) / B, B, 0, stream>>>(x2, Wc3, bc3, out, HID / 2, OUT_DIM, 0);
}
